// EPMoE_W4A8_45329084842370
// MI455X (gfx1250) — compile-verified, matrix operand order checked by reading, not run on hardware
//
#include <hip/hip_runtime.h>
#include <hip/hip_fp16.h>
#include <math.h>
#include <stdint.h>

// Problem constants (from reference): T=2048, H=2048, I=1408, E=32, K=6
#define T_TOK 2048
#define HDIM  2048
#define IDIM  1408
#define NEXP  32
#define TOPK  6
#define PTOT  (T_TOK * TOPK)           // 12288 expanded rows
#define ALIGN 32                       // expert group padding (one supertile)
#define PPAD  (PTOT + NEXP * ALIGN)    // 13312 rows worst case
#define NSUPER (PPAD / 32)             // 416 row supertiles of 32

typedef __attribute__((ext_vector_type(8))) int v8i;

// ---------------------------------------------------------------------------
// Load one 8-bit WMMA operand fragment (16x64 A, or B given as 16 N-major
// rows x 64 K). Per ISA 7.12.2: lanes 0-15 hold row m, K={0-7,16-23,32-39,48-55};
// lanes 16-31 hold row m with K offset +8. base = first row of the 16-row block,
// ld = row stride bytes, k0 = K offset (multiple of 64).
// ---------------------------------------------------------------------------
__device__ __forceinline__ v8i load_frag8(const int8_t* __restrict__ base,
                                          int ld, int lane, int k0) {
  int r = lane & 15;
  int o = (lane < 16) ? 0 : 8;
  const int8_t* p = base + (size_t)r * ld + k0 + o;
  int2 a = *(const int2*)(p + 0);
  int2 b = *(const int2*)(p + 16);
  int2 c = *(const int2*)(p + 32);
  int2 d = *(const int2*)(p + 48);
  v8i f;
  f[0] = a.x; f[1] = a.y; f[2] = b.x; f[3] = b.y;
  f[4] = c.x; f[5] = c.y; f[6] = d.x; f[7] = d.y;
  return f;
}

#define WMMA_IU8(A, B, C) \
  __builtin_amdgcn_wmma_i32_16x16x64_iu8(true, (A), true, (B), (C), false, false)

// ---------------------------------------------------------------------------
// 0) init: zero output + counters, poison row maps
// ---------------------------------------------------------------------------
__global__ void k_init(float* __restrict__ out, int* __restrict__ counts,
                       int* __restrict__ fill, int* __restrict__ tile_expert,
                       int* __restrict__ sorted_tok) {
  int idx = blockIdx.x * blockDim.x + threadIdx.x;
  if (idx < T_TOK * HDIM) out[idx] = 0.0f;
  if (idx < NEXP) { counts[idx] = 0; fill[idx] = 0; }
  if (idx < NSUPER) tile_expert[idx] = -1;
  if (idx < PPAD) sorted_tok[idx] = -1;
}

// ---------------------------------------------------------------------------
// 1) router: softmax + top-6 (ties -> lowest index, like lax.top_k) + renorm
// ---------------------------------------------------------------------------
__global__ void k_router(const float* __restrict__ logits,
                         int* __restrict__ topk_ids, float* __restrict__ topk_w,
                         int* __restrict__ counts) {
  int t = blockIdx.x * blockDim.x + threadIdx.x;
  if (t >= T_TOK) return;
  float l[NEXP];
  float m = -INFINITY;
  for (int e = 0; e < NEXP; ++e) { l[e] = logits[t * NEXP + e]; m = fmaxf(m, l[e]); }
  for (int e = 0; e < NEXP; ++e) l[e] = __expf(l[e] - m);  // unnormalized probs
  bool used[NEXP];
  for (int e = 0; e < NEXP; ++e) used[e] = false;
  int ids[TOPK]; float w[TOPK]; float wsum = 0.0f;
  for (int k = 0; k < TOPK; ++k) {
    float best = -1.0f; int bi = 0;
    for (int e = 0; e < NEXP; ++e)
      if (!used[e] && l[e] > best) { best = l[e]; bi = e; }
    used[bi] = true; ids[k] = bi; w[k] = best; wsum += best;
  }
  float inv = 1.0f / wsum;
  for (int k = 0; k < TOPK; ++k) {
    topk_ids[t * TOPK + k] = ids[k];
    topk_w[t * TOPK + k] = w[k] * inv;
    atomicAdd(&counts[ids[k]], 1);
  }
}

// ---------------------------------------------------------------------------
// 2) scan: 32-aligned exclusive prefix of counts, fill supertile->expert map
// ---------------------------------------------------------------------------
__global__ void k_scan(const int* __restrict__ counts, int* __restrict__ pad_off,
                       int* __restrict__ tile_expert) {
  __shared__ int offs[NEXP + 1];
  if (threadIdx.x == 0) {
    int off = 0;
    for (int e = 0; e < NEXP; ++e) {
      offs[e] = off;
      off += ((counts[e] + 31) >> 5) << 5;
    }
    offs[NEXP] = off;
    for (int e = 0; e <= NEXP; ++e) pad_off[e] = offs[e];
  }
  __syncthreads();
  int e = threadIdx.x;
  if (e < NEXP)
    for (int tt = offs[e] >> 5; tt < (offs[e + 1] >> 5); ++tt) tile_expert[tt] = e;
}

// ---------------------------------------------------------------------------
// 3) scatter: place each (token, k) entry into its expert's padded segment
// ---------------------------------------------------------------------------
__global__ void k_scatter(const int* __restrict__ topk_ids,
                          const float* __restrict__ topk_w,
                          const int* __restrict__ pad_off, int* __restrict__ fill,
                          int* __restrict__ sorted_tok, float* __restrict__ sorted_w) {
  int idx = blockIdx.x * blockDim.x + threadIdx.x;
  if (idx >= PTOT) return;
  int e = topk_ids[idx];
  int pos = pad_off[e] + atomicAdd(&fill[e], 1);
  sorted_tok[pos] = idx / TOPK;
  sorted_w[pos] = topk_w[idx];
}

// ---------------------------------------------------------------------------
// 4) quantize gathered hidden rows -> int8 (per-row absmax / 127)
// ---------------------------------------------------------------------------
__global__ void k_quant_x(const float* __restrict__ hidden,
                          const int* __restrict__ sorted_tok,
                          int8_t* __restrict__ Xq, float* __restrict__ xs) {
  int row = blockIdx.x;
  int tid = threadIdx.x;  // 256 threads
  __shared__ float red[8];
  int tok = sorted_tok[row];
  int8_t* qrow = Xq + (size_t)row * HDIM;
  if (tok < 0) {
    int4 z; z.x = 0; z.y = 0; z.z = 0; z.w = 0;
    for (int i = tid; i < HDIM / 16; i += 256) ((int4*)qrow)[i] = z;
    if (tid == 0) xs[row] = 0.0f;
    return;
  }
  const float* x = hidden + (size_t)tok * HDIM;
  float m = 0.0f;
  for (int i = tid; i < HDIM; i += 256) m = fmaxf(m, fabsf(x[i]));
  for (int off = 16; off > 0; off >>= 1) m = fmaxf(m, __shfl_xor(m, off, 32));
  if ((tid & 31) == 0) red[tid >> 5] = m;
  __syncthreads();
  m = red[0];
  for (int i = 1; i < 8; ++i) m = fmaxf(m, red[i]);
  float s = fmaxf(m, 1e-6f) / 127.0f;
  float inv = 1.0f / s;
  for (int i = tid; i < HDIM; i += 256) {
    float q = rintf(x[i] * inv);               // RNE, matches jnp.round
    q = fminf(fmaxf(q, -127.0f), 127.0f);
    qrow[i] = (int8_t)q;
  }
  if (tid == 0) xs[row] = s;
}

// ---------------------------------------------------------------------------
// GEMM1 epilogue for one 16-row block: dequant gate/up, silu, store fp16
// ---------------------------------------------------------------------------
__device__ __forceinline__ void epi_silu(v8i cg0, v8i cg1, v8i cu0, v8i cu1,
                                         int row0, int n, int col0,
                                         const float* __restrict__ sg,
                                         const float* __restrict__ xs,
                                         _Float16* __restrict__ hmid) {
#pragma unroll
  for (int r = 0; r < 8; ++r) {
    int row = row0 + r;
    float s = xs[row];
    float g0 = (float)cg0[r] * s * sg[col0 + n];
    float g1 = (float)cg1[r] * s * sg[col0 + 16 + n];
    float u0 = (float)cu0[r] * s * sg[IDIM + col0 + n];
    float u1 = (float)cu1[r] * s * sg[IDIM + col0 + 16 + n];
    float h0 = (g0 / (1.0f + __expf(-g0))) * u0;  // silu(g)*u
    float h1 = (g1 / (1.0f + __expf(-g1))) * u1;
    hmid[(size_t)row * IDIM + col0 + n]      = (_Float16)h0;
    hmid[(size_t)row * IDIM + col0 + 16 + n] = (_Float16)h1;
  }
}

// ---------------------------------------------------------------------------
// 5) GEMM1 (int8 WMMA) fused with SiLU(gate)*up -> hmid (fp16)
//    grid: (NSUPER, 6), block 256 = 8 waves.
//    Each wave: 32 rows x (32 gate cols + 32 up cols) = 8 accumulators,
//    6 fragment loads / 8 WMMAs per K-step. Latency is hidden by wave-level
//    parallelism; fragments are single-buffered to avoid spills.
// ---------------------------------------------------------------------------
__global__ void __launch_bounds__(256, 1)
k_gemm1(const int8_t* __restrict__ Xq, const int8_t* __restrict__ w13,
        const float* __restrict__ xs, const float* __restrict__ w13_scale,
        const int* __restrict__ tile_expert, _Float16* __restrict__ hmid) {
  int super = blockIdx.x;
  int e = tile_expert[super];
  if (e < 0) return;
  int wave = threadIdx.x >> 5;
  int lane = threadIdx.x & 31;
  int col0 = blockIdx.y * 256 + wave * 32;   // gate/hmid column base
  if (col0 >= IDIM) return;                  // IDIM % 32 == 0

  const int8_t* arow0 = Xq + (size_t)super * 32 * HDIM;
  const int8_t* arow1 = arow0 + (size_t)16 * HDIM;
  const int8_t* wbase = w13 + (size_t)e * 2 * IDIM * HDIM;
  const int8_t* bg0p = wbase + (size_t)col0 * HDIM;
  const int8_t* bg1p = bg0p + (size_t)16 * HDIM;
  const int8_t* bu0p = wbase + (size_t)(IDIM + col0) * HDIM;
  const int8_t* bu1p = bu0p + (size_t)16 * HDIM;

  // accumulators: [A-block m][gate0, gate1, up0, up1]
  v8i cg0_0 = {}, cg1_0 = {}, cu0_0 = {}, cu1_0 = {};
  v8i cg0_1 = {}, cg1_1 = {}, cu0_1 = {}, cu1_1 = {};

  for (int k = 0; k < HDIM; k += 64) {
    v8i a0 = load_frag8(arow0, HDIM, lane, k);
    v8i a1 = load_frag8(arow1, HDIM, lane, k);
    v8i g0 = load_frag8(bg0p, HDIM, lane, k);
    v8i g1 = load_frag8(bg1p, HDIM, lane, k);
    v8i u0 = load_frag8(bu0p, HDIM, lane, k);
    v8i u1 = load_frag8(bu1p, HDIM, lane, k);

    cg0_0 = WMMA_IU8(a0, g0, cg0_0);
    cg1_0 = WMMA_IU8(a0, g1, cg1_0);
    cu0_0 = WMMA_IU8(a0, u0, cu0_0);
    cu1_0 = WMMA_IU8(a0, u1, cu1_0);
    cg0_1 = WMMA_IU8(a1, g0, cg0_1);
    cg1_1 = WMMA_IU8(a1, g1, cg1_1);
    cu0_1 = WMMA_IU8(a1, u0, cu0_1);
    cu1_1 = WMMA_IU8(a1, u1, cu1_1);
  }

  // Epilogue: C/D layout -> vgpr r holds (row = r + (lane<16?0:8), col = lane&15)
  int n = lane & 15;
  int rb = (lane < 16) ? 0 : 8;
  const float* sg = w13_scale + (size_t)e * 2 * IDIM;
  epi_silu(cg0_0, cg1_0, cu0_0, cu1_0, super * 32 + rb,      n, col0, sg, xs, hmid);
  epi_silu(cg0_1, cg1_1, cu0_1, cu1_1, super * 32 + 16 + rb, n, col0, sg, xs, hmid);
}

// ---------------------------------------------------------------------------
// 6) quantize hmid rows -> int8
// ---------------------------------------------------------------------------
__global__ void k_quant_h(const _Float16* __restrict__ hmid,
                          const int* __restrict__ tile_expert,
                          int8_t* __restrict__ Hq, float* __restrict__ hs) {
  int row = blockIdx.x;
  int tid = threadIdx.x;
  __shared__ float red[8];
  int8_t* qrow = Hq + (size_t)row * IDIM;
  if (tile_expert[row >> 5] < 0) {
    int4 z; z.x = 0; z.y = 0; z.z = 0; z.w = 0;
    for (int i = tid; i < IDIM / 16; i += 256) ((int4*)qrow)[i] = z;
    if (tid == 0) hs[row] = 0.0f;
    return;
  }
  const _Float16* x = hmid + (size_t)row * IDIM;
  float m = 0.0f;
  for (int i = tid; i < IDIM; i += 256) m = fmaxf(m, fabsf((float)x[i]));
  for (int off = 16; off > 0; off >>= 1) m = fmaxf(m, __shfl_xor(m, off, 32));
  if ((tid & 31) == 0) red[tid >> 5] = m;
  __syncthreads();
  m = red[0];
  for (int i = 1; i < 8; ++i) m = fmaxf(m, red[i]);
  float s = fmaxf(m, 1e-6f) / 127.0f;
  float inv = 1.0f / s;
  for (int i = tid; i < IDIM; i += 256) {
    float q = rintf((float)x[i] * inv);
    q = fminf(fmaxf(q, -127.0f), 127.0f);
    qrow[i] = (int8_t)q;
  }
  if (tid == 0) hs[row] = s;
}

// ---------------------------------------------------------------------------
// GEMM2 epilogue for one 16x16 acc tile: dequant + weighted atomic scatter
// ---------------------------------------------------------------------------
__device__ __forceinline__ void epi_scatter(v8i c, int row0, int col, int n,
                                            const float* __restrict__ sc,
                                            const float* __restrict__ hs,
                                            const int* __restrict__ sorted_tok,
                                            const float* __restrict__ sorted_w,
                                            float* __restrict__ out) {
#pragma unroll
  for (int r = 0; r < 8; ++r) {
    int row = row0 + r;
    int tok = sorted_tok[row];
    if (tok < 0) continue;
    float f = hs[row] * sorted_w[row];
    atomicAdd(&out[(size_t)tok * HDIM + col + n], (float)c[r] * f * sc[col + n]);
  }
}

// ---------------------------------------------------------------------------
// 7) GEMM2 (int8 WMMA) + weighted scatter-add into out
//    grid: (NSUPER, 4), block 256 = 8 waves.
//    Each wave: 32 rows x 64 H cols = 8 accumulators, single-buffered.
// ---------------------------------------------------------------------------
__global__ void __launch_bounds__(256, 1)
k_gemm2(const int8_t* __restrict__ Hq, const int8_t* __restrict__ w2,
        const float* __restrict__ hs, const float* __restrict__ w2_scale,
        const int* __restrict__ tile_expert,
        const int* __restrict__ sorted_tok,
        const float* __restrict__ sorted_w, float* __restrict__ out) {
  int super = blockIdx.x;
  int e = tile_expert[super];
  if (e < 0) return;
  int wave = threadIdx.x >> 5;
  int lane = threadIdx.x & 31;
  int col0 = blockIdx.y * 512 + wave * 64;   // H column base; 4*512 == HDIM

  const int8_t* arow0 = Hq + (size_t)super * 32 * IDIM;
  const int8_t* arow1 = arow0 + (size_t)16 * IDIM;
  const int8_t* bbase = w2 + (size_t)e * HDIM * IDIM + (size_t)col0 * IDIM;
  const int8_t* b0p = bbase;
  const int8_t* b1p = bbase + (size_t)16 * IDIM;
  const int8_t* b2p = bbase + (size_t)32 * IDIM;
  const int8_t* b3p = bbase + (size_t)48 * IDIM;

  v8i c0_0 = {}, c1_0 = {}, c2_0 = {}, c3_0 = {};
  v8i c0_1 = {}, c1_1 = {}, c2_1 = {}, c3_1 = {};

  for (int k = 0; k < IDIM; k += 64) {
    v8i a0 = load_frag8(arow0, IDIM, lane, k);
    v8i a1 = load_frag8(arow1, IDIM, lane, k);
    v8i b0 = load_frag8(b0p, IDIM, lane, k);
    v8i b1 = load_frag8(b1p, IDIM, lane, k);
    v8i b2 = load_frag8(b2p, IDIM, lane, k);
    v8i b3 = load_frag8(b3p, IDIM, lane, k);

    c0_0 = WMMA_IU8(a0, b0, c0_0);
    c1_0 = WMMA_IU8(a0, b1, c1_0);
    c2_0 = WMMA_IU8(a0, b2, c2_0);
    c3_0 = WMMA_IU8(a0, b3, c3_0);
    c0_1 = WMMA_IU8(a1, b0, c0_1);
    c1_1 = WMMA_IU8(a1, b1, c1_1);
    c2_1 = WMMA_IU8(a1, b2, c2_1);
    c3_1 = WMMA_IU8(a1, b3, c3_1);
  }

  int n = lane & 15;
  int rb = (lane < 16) ? 0 : 8;
  const float* sc = w2_scale + (size_t)e * HDIM;
  int r0 = super * 32 + rb;
  int r1 = super * 32 + 16 + rb;
  epi_scatter(c0_0, r0, col0 +  0, n, sc, hs, sorted_tok, sorted_w, out);
  epi_scatter(c1_0, r0, col0 + 16, n, sc, hs, sorted_tok, sorted_w, out);
  epi_scatter(c2_0, r0, col0 + 32, n, sc, hs, sorted_tok, sorted_w, out);
  epi_scatter(c3_0, r0, col0 + 48, n, sc, hs, sorted_tok, sorted_w, out);
  epi_scatter(c0_1, r1, col0 +  0, n, sc, hs, sorted_tok, sorted_w, out);
  epi_scatter(c1_1, r1, col0 + 16, n, sc, hs, sorted_tok, sorted_w, out);
  epi_scatter(c2_1, r1, col0 + 32, n, sc, hs, sorted_tok, sorted_w, out);
  epi_scatter(c3_1, r1, col0 + 48, n, sc, hs, sorted_tok, sorted_w, out);
}

// ---------------------------------------------------------------------------
// Host: workspace carve-out + launch sequence
// ---------------------------------------------------------------------------
extern "C" void kernel_launch(void* const* d_in, const int* in_sizes, int n_in,
                              void* d_out, int out_size, void* d_ws, size_t ws_size,
                              hipStream_t stream) {
  const float*  hidden    = (const float*)d_in[0];
  const float*  logits    = (const float*)d_in[1];
  const int8_t* w13       = (const int8_t*)d_in[2];
  const int8_t* w2        = (const int8_t*)d_in[3];
  const float*  w13_scale = (const float*)d_in[4];
  const float*  w2_scale  = (const float*)d_in[5];
  float* out = (float*)d_out;

  uint8_t* ws = (uint8_t*)d_ws;
  size_t off = 0;
  auto carve = [&](size_t bytes) -> void* {
    void* p = ws + off;
    off = (off + bytes + 255) & ~(size_t)255;
    return p;
  };
  int8_t*   Xq          = (int8_t*)  carve((size_t)PPAD * HDIM);
  int8_t*   Hq          = (int8_t*)  carve((size_t)PPAD * IDIM);
  _Float16* Hmid        = (_Float16*)carve((size_t)PPAD * IDIM * 2);
  float*    xs          = (float*)   carve((size_t)PPAD * 4);
  float*    hs          = (float*)   carve((size_t)PPAD * 4);
  int*      sorted_tok  = (int*)     carve((size_t)PPAD * 4);
  float*    sorted_w    = (float*)   carve((size_t)PPAD * 4);
  int*      tile_expert = (int*)     carve((size_t)NSUPER * 4);
  int*      counts      = (int*)     carve((size_t)NEXP * 4);
  int*      pad_off     = (int*)     carve((size_t)(NEXP + 1) * 4);
  int*      fill        = (int*)     carve((size_t)NEXP * 4);
  int*      topk_ids    = (int*)     carve((size_t)PTOT * 4);
  float*    topk_w      = (float*)   carve((size_t)PTOT * 4);

  k_init<<<(T_TOK * HDIM + 255) / 256, 256, 0, stream>>>(out, counts, fill,
                                                         tile_expert, sorted_tok);
  k_router<<<(T_TOK + 255) / 256, 256, 0, stream>>>(logits, topk_ids, topk_w, counts);
  k_scan<<<1, 32, 0, stream>>>(counts, pad_off, tile_expert);
  k_scatter<<<(PTOT + 255) / 256, 256, 0, stream>>>(topk_ids, topk_w, pad_off, fill,
                                                    sorted_tok, sorted_w);
  k_quant_x<<<PPAD, 256, 0, stream>>>(hidden, sorted_tok, Xq, xs);
  k_gemm1<<<dim3(NSUPER, 6), 256, 0, stream>>>(Xq, w13, xs, w13_scale,
                                               tile_expert, Hmid);
  k_quant_h<<<PPAD, 256, 0, stream>>>(Hmid, tile_expert, Hq, hs);
  k_gemm2<<<dim3(NSUPER, 4), 256, 0, stream>>>(Hq, w2, hs, w2_scale, tile_expert,
                                               sorted_tok, sorted_w, out);
}
